// Relative_Attention_inner_40183714021648
// MI455X (gfx1250) — matrix operand order repaired and checked by reading, not compiled, not run on hardware
//
#include <hip/hip_runtime.h>
#include <hip/hip_bf16.h>

// Shapes from the reference: B=4, H=16, L=1024, D=64; transpose==1 path only.
#define LL 1024
#define DD 64
#define BH 64
#define SLAB 256   // l-slab: out slab = 64*256*1024*4 = 64MB, L2-resident (192MB)

typedef __attribute__((ext_vector_type(16))) __bf16          v16bf;
typedef __attribute__((ext_vector_type(16))) unsigned short  v16us;
typedef __attribute__((ext_vector_type(8)))  unsigned short  v8us;
typedef __attribute__((ext_vector_type(8)))  float           v8f;
typedef __attribute__((ext_vector_type(4)))  float           v4f;

// Split f32 into bf16 hi (hardware RNE cvt) and bf16 lo (cvt of residual).
__device__ __forceinline__ void split2(float f, unsigned short& h, unsigned short& l) {
  __bf16 hb = (__bf16)f;
  unsigned short hu = __builtin_bit_cast(unsigned short, hb);
  float fh = __builtin_bit_cast(float, (unsigned int)hu << 16);
  __bf16 lb = (__bf16)(f - fh);
  h = hu;
  l = __builtin_bit_cast(unsigned short, lb);
}

__device__ __forceinline__ v8f wmma_bf16(v16us a, v16us b, v8f c) {
  v16bf av = __builtin_bit_cast(v16bf, a);
  v16bf bv = __builtin_bit_cast(v16bf, b);
  return __builtin_amdgcn_wmma_f32_16x16x32_bf16(false, av, false, bv,
                                                 (short)0, c, false, false);
}

// Load a 16-element bf16 fragment (two 16B chunks) from LDS.
__device__ __forceinline__ v16us frag_2x8(const unsigned short* p0,
                                          const unsigned short* p1) {
  v8us a = *(const v8us*)p0;
  v8us b = *(const v8us*)p1;
  return __builtin_shufflevector(a, b, 0,1,2,3,4,5,6,7,8,9,10,11,12,13,14,15);
}

// ---------------------------------------------------------------------------
// Kernel 1:  out[bh,l,m] = sum_d x[bh,l,d] * y[bh,m,d]   for l in [l0,l0+SLAB)
// Grid (SLAB/128, L/128, BH); block = 256 threads = 8 waves (2x4 wave grid),
// each wave owns a 64x32 sub-tile = 4x2 WMMA 16x16 accumulators.
// A(128x64) and B(128x64) panels are converted to split-bf16 once per block
// into LDS; waves fetch fragments with ds_load_b128. Stores stay temporal:
// k_xtz re-reads this slab from L2.
// ---------------------------------------------------------------------------
__global__ __launch_bounds__(256) void k_xy(const float* __restrict__ x,
                                            const float* __restrict__ y,
                                            float* __restrict__ out,
                                            int l0) {
  const int bh   = blockIdx.z;
  const int m0   = l0 + blockIdx.x * 128;  // l tile (rows, from x)
  const int n0   = blockIdx.y * 128;       // m tile (cols, from y)
  const int tid  = threadIdx.x;

  __shared__ unsigned short sA_h[128][64];   // 16 KB
  __shared__ unsigned short sA_l[128][64];   // 16 KB
  __shared__ unsigned short sB_h[128][64];   // 16 KB
  __shared__ unsigned short sB_l[128][64];   // 16 KB

  const float* xb = x + (size_t)bh * (LL * DD);
  const float* yb = y + (size_t)bh * (LL * DD);

  // Cooperative stage + split: each thread handles 32 floats of A and 32 of B.
  {
    const int row = tid >> 1;              // 0..127
    const int c0  = (tid & 1) * 32;        // 0 or 32
    const float* pa = xb + (size_t)(m0 + row) * DD + c0;
    const float* pb = yb + (size_t)(n0 + row) * DD + c0;
    #pragma unroll
    for (int c = 0; c < 8; ++c) {
      v4f qa = *(const v4f*)(pa + c * 4);
      v4f qb = *(const v4f*)(pb + c * 4);
      #pragma unroll
      for (int e = 0; e < 4; ++e) {
        unsigned short hh, ll;
        split2(qa[e], hh, ll);
        sA_h[row][c0 + c * 4 + e] = hh;
        sA_l[row][c0 + c * 4 + e] = ll;
        split2(qb[e], hh, ll);
        sB_h[row][c0 + c * 4 + e] = hh;
        sB_l[row][c0 + c * 4 + e] = ll;
      }
    }
  }
  __syncthreads();

  const int w    = tid >> 5;
  const int lane = tid & 31;
  const int half = lane >> 4;
  const int l16  = lane & 15;
  const int wm   = w >> 2;                 // 0..1
  const int wn   = w & 3;                  // 0..3

  v8f acc[4][2];
  #pragma unroll
  for (int i = 0; i < 4; ++i)
    #pragma unroll
    for (int j = 0; j < 2; ++j) acc[i][j] = (v8f)0.0f;

  #pragma unroll
  for (int ks = 0; ks < DD; ks += 32) {
    const int kb = ks + half * 8;          // A chunk base
    const int k0 = ks + half * 8;          // B chunk base

    v16us a_hi[4], a_lo[4];
    #pragma unroll
    for (int i = 0; i < 4; ++i) {
      const int row = wm * 64 + i * 16 + l16;
      a_hi[i] = frag_2x8(&sA_h[row][kb], &sA_h[row][kb + 16]);
      a_lo[i] = frag_2x8(&sA_l[row][kb], &sA_l[row][kb + 16]);
    }
    v16us b_hi[2], b_lo[2];
    #pragma unroll
    for (int j = 0; j < 2; ++j) {
      const int col = wn * 32 + j * 16 + l16;
      b_hi[j] = frag_2x8(&sB_h[col][k0], &sB_h[col][k0 + 16]);
      b_lo[j] = frag_2x8(&sB_l[col][k0], &sB_l[col][k0 + 16]);
    }

    // Split-precision product: x*y ~= hi*hi + lo*hi + hi*lo
    #pragma unroll
    for (int i = 0; i < 4; ++i)
      #pragma unroll
      for (int j = 0; j < 2; ++j) {
        acc[i][j] = wmma_bf16(a_hi[i], b_hi[j], acc[i][j]);
        acc[i][j] = wmma_bf16(a_lo[i], b_hi[j], acc[i][j]);
        acc[i][j] = wmma_bf16(a_hi[i], b_lo[j], acc[i][j]);
      }
  }

  // C/D layout: VGPR r -> row (r + half*8), col l16
  #pragma unroll
  for (int i = 0; i < 4; ++i) {
    const int rowBase = m0 + wm * 64 + i * 16 + half * 8;
    #pragma unroll
    for (int j = 0; j < 2; ++j) {
      const int col = n0 + wn * 32 + j * 16 + l16;
      #pragma unroll
      for (int r = 0; r < 8; ++r)
        out[((size_t)bh * LL + (rowBase + r)) * LL + col] = acc[i][j][r];
    }
  }
}

// ---------------------------------------------------------------------------
// Kernel 2:  out[bh,l,m] += sum_d x[bh,l,d] * z[l,m,d]   for l in [l0,l0+SLAB)
// Per l: C[64(bh) x 1024(m)] = X_l[64x64] @ Z_l^T.  Grid (L/128, SLAB).
// X_l staged in LDS pre-split to bf16 hi/lo; each wave owns a 64x16 strip.
// z is streamed exactly once -> non-temporal loads (protect L2 out-slab).
// Final stores are never re-read -> non-temporal stores.
// ---------------------------------------------------------------------------
__global__ __launch_bounds__(256) void k_xtz(const float* __restrict__ x,
                                             const float* __restrict__ z,
                                             float* __restrict__ out,
                                             int l0) {
  const int l  = l0 + blockIdx.y;
  const int n0 = blockIdx.x * 128;

  __shared__ unsigned short sh_hi[64][64];  // [bh][d], 8 KB
  __shared__ unsigned short sh_lo[64][64];  // 8 KB

  const int tid = threadIdx.x;
  {
    const int row = tid >> 2;
    const int c0  = (tid & 3) * 16;
    const float* p = x + ((size_t)row * LL + l) * DD + c0;
    #pragma unroll
    for (int c = 0; c < 4; ++c) {
      v4f q = *(const v4f*)(p + c * 4);
      #pragma unroll
      for (int e = 0; e < 4; ++e) {
        unsigned short hh, ll;
        split2(q[e], hh, ll);
        sh_hi[row][c0 + c * 4 + e] = hh;
        sh_lo[row][c0 + c * 4 + e] = ll;
      }
    }
  }
  __syncthreads();

  const int w    = tid >> 5;
  const int lane = tid & 31;
  const int half = lane >> 4;
  const int l16  = lane & 15;
  const int n    = n0 + w * 16 + l16;      // output m, z row

  v8f acc[4];
  #pragma unroll
  for (int i = 0; i < 4; ++i) acc[i] = (v8f)0.0f;

  #pragma unroll
  for (int ks = 0; ks < DD; ks += 32) {
    // B fragment from z: 16 contiguous f32 at z[l][n][ks+half*16 ..] (NT loads)
    v16us zb_h, zb_l;
    {
      const float* p = z + ((size_t)l * LL + n) * DD + ks + half * 16;
      v4f q0 = __builtin_nontemporal_load((const v4f*)(p + 0));
      v4f q1 = __builtin_nontemporal_load((const v4f*)(p + 4));
      v4f q2 = __builtin_nontemporal_load((const v4f*)(p + 8));
      v4f q3 = __builtin_nontemporal_load((const v4f*)(p + 12));
      #pragma unroll
      for (int e = 0; e < 4; ++e) {
        unsigned short hh, ll;
        split2(q0[e], hh, ll); zb_h[e]      = hh; zb_l[e]      = ll;
        split2(q1[e], hh, ll); zb_h[4 + e]  = hh; zb_l[4 + e]  = ll;
        split2(q2[e], hh, ll); zb_h[8 + e]  = hh; zb_l[8 + e]  = ll;
        split2(q3[e], hh, ll); zb_h[12 + e] = hh; zb_l[12 + e] = ll;
      }
    }
    #pragma unroll
    for (int i = 0; i < 4; ++i) {
      const int row = i * 16 + l16;
      const int kb  = ks + half * 16;
      v16us ah = frag_2x8(&sh_hi[row][kb], &sh_hi[row][kb + 8]);
      v16us al = frag_2x8(&sh_lo[row][kb], &sh_lo[row][kb + 8]);
      acc[i] = wmma_bf16(ah, zb_h, acc[i]);
      acc[i] = wmma_bf16(al, zb_h, acc[i]);
      acc[i] = wmma_bf16(ah, zb_l, acc[i]);
    }
  }

  // RMW accumulate into out: the read hits L2 (same slab just written by k_xy);
  // the final store is non-temporal (never re-read).
  #pragma unroll
  for (int i = 0; i < 4; ++i) {
    const int bhBase = i * 16 + half * 8;
    #pragma unroll
    for (int r = 0; r < 8; ++r) {
      size_t idx = ((size_t)(bhBase + r) * LL + l) * LL + n;
      float old = out[idx];
      __builtin_nontemporal_store(old + acc[i][r], &out[idx]);
    }
  }
}

extern "C" void kernel_launch(void* const* d_in, const int* in_sizes, int n_in,
                              void* d_out, int out_size, void* d_ws, size_t ws_size,
                              hipStream_t stream) {
  const float* x = (const float*)d_in[0];
  const float* y = (const float*)d_in[1];
  const float* z = (const float*)d_in[2];
  float* out = (float*)d_out;

  // Interleave the two phases over l-slabs so the intermediate out-slab (64MB)
  // is still L2-resident (192MB) when k_xtz performs its read-modify-write.
  for (int l0 = 0; l0 < LL; l0 += SLAB) {
    dim3 g1(SLAB / 128, LL / 128, BH);   // 2 x 8 x 64
    k_xy<<<g1, 256, 0, stream>>>(x, y, out, l0);
    dim3 g2(LL / 128, SLAB);             // 8 x 256
    k_xtz<<<g2, 256, 0, stream>>>(x, z, out, l0);
  }
}
